// DI_GNN_15109694947989
// MI455X (gfx1250) — matrix operand order repaired and checked by reading, not compiled, not run on hardware
//
#include <hip/hip_runtime.h>
#include <stdint.h>

#define Bg   128
#define Nn   512
#define Kn   6
#define NBn  (Bg * Nn)      // 65536 nodes
#define Hf   256
#define INFt 128            // IN_FEAT
#define EPS_BN 1e-5f

typedef __attribute__((ext_vector_type(16))) __bf16 bf16x16;
typedef __attribute__((ext_vector_type(8)))  __bf16 bf16x8;
typedef __attribute__((ext_vector_type(8)))  float  f32x8;

__device__ __forceinline__ __bf16 f2bf(float f) {
    union { float f; unsigned u; } a; a.f = f;
    unsigned r = a.u + 0x7FFFu + ((a.u >> 16) & 1u);   // round-to-nearest-even
    union { unsigned short s; __bf16 b; } o; o.s = (unsigned short)(r >> 16);
    return o.b;
}

__device__ __forceinline__ bf16x16 load_frag2(const __bf16* p0, const __bf16* p1) {
    bf16x8 lo = *reinterpret_cast<const bf16x8*>(p0);
    bf16x8 hi = *reinterpret_cast<const bf16x8*>(p1);
    bf16x16 r;
#pragma unroll
    for (int e = 0; e < 8; ++e) { r[e] = lo[e]; r[e + 8] = hi[e]; }
    return r;
}

// ---------------- conversions ----------------
__global__ __launch_bounds__(256)
void cvt_f32_bf16(const float* __restrict__ in, __bf16* __restrict__ out, int n) {
    int i = blockIdx.x * 256 + threadIdx.x;
    if (i < n) out[i] = f2bf(in[i]);
}

// W [Kd, Nw] row-major f32  ->  Wt [Nw, Kd] row-major bf16
__global__ __launch_bounds__(256)
void cvt_transpose_bf16(const float* __restrict__ W, __bf16* __restrict__ Wt, int Kd, int Nw) {
    int i = blockIdx.x * 256 + threadIdx.x;
    if (i < Kd * Nw) {
        int k = i / Nw, n = i % Nw;
        Wt[(size_t)n * Kd + k] = f2bf(W[i]);
    }
}

// ---------------- kNN graph (per-graph 512x512, 3D) ----------------
__global__ __launch_bounds__(512)
void knn_kernel(const float* __restrict__ pos, int* __restrict__ src) {
    __shared__ float sp[Nn * 3];
    const int b = blockIdx.x, tid = threadIdx.x;
    const int base = b * Nn;
    sp[tid * 3 + 0] = pos[(size_t)(base + tid) * 3 + 0];
    sp[tid * 3 + 1] = pos[(size_t)(base + tid) * 3 + 1];
    sp[tid * 3 + 2] = pos[(size_t)(base + tid) * 3 + 2];
    __syncthreads();
    const float px = sp[tid * 3 + 0], py = sp[tid * 3 + 1], pz = sp[tid * 3 + 2];
    float bd[Kn]; int bi[Kn];
#pragma unroll
    for (int k = 0; k < Kn; ++k) { bd[k] = 3.4e38f; bi[k] = 0; }
    for (int j = 0; j < Nn; ++j) {
        if (j == tid) continue;
        float dx = px - sp[j * 3 + 0];
        float dy = py - sp[j * 3 + 1];
        float dz = pz - sp[j * 3 + 2];
        float d2 = dx * dx + dy * dy + dz * dz;
        if (d2 < bd[Kn - 1]) {
            int p = Kn - 1;
            while (p > 0 && d2 < bd[p - 1]) { bd[p] = bd[p - 1]; bi[p] = bi[p - 1]; --p; }
            bd[p] = d2; bi[p] = j;
        }
    }
#pragma unroll
    for (int k = 0; k < Kn; ++k)
        src[(size_t)(base + tid) * Kn + k] = base + bi[k];
}

// ---------------- bf16 WMMA GEMM: C[M,Ncols] = A[M,Ka] * Bt[Ncols,Ka]^T ----------------
// Wave tile: 32(M) x 64(N). B fragments shared across both 16-row A groups.
// Optional fp32 epilogue: C += pos[M,3] @ posW[3,Ncols]   (exact handling of W1 rows 128..130)
__global__ __launch_bounds__(256)
void gemm_bf16_wmma(const __bf16* __restrict__ A, const __bf16* __restrict__ Bt,
                    float* __restrict__ C, int M, int Ka, int Ncols,
                    const float* __restrict__ posW, const float* __restrict__ pos) {
    const int wid  = blockIdx.x * (blockDim.x >> 5) + (threadIdx.x >> 5);
    const int lane = threadIdx.x & 31;
    const int ntN  = Ncols >> 6;                 // N tiles of 64
    const int mt = wid / ntN, nt = wid % ntN;    // mt: 32-row tile
    if (mt * 32 >= M) return;                    // wave-uniform
    const int half = lane >> 4, l15 = lane & 15;
    const __bf16* arow0 = A + (size_t)(mt * 32 + l15) * Ka;        // rows 0..15 of tile
    const __bf16* arow1 = A + (size_t)(mt * 32 + 16 + l15) * Ka;   // rows 16..31 of tile
    const int nbase = nt * 64;

    f32x8 acc[2][4] = {};
    for (int kb = 0; kb < Ka; kb += 32) {
        // ---- issue ALL loads for this k-step first (waits amortize over 8 WMMAs) ----
        // A fragments (16x32 bf16): lanes 0-15 K{0..7,16..23}; lanes 16-31 K{8..15,24..31}
        bf16x16 a0 = load_frag2(arow0 + kb + half * 8, arow0 + kb + 16 + half * 8);
        bf16x16 a1 = load_frag2(arow1 + kb + half * 8, arow1 + kb + 16 + half * 8);
        // B fragments (32x16 bf16): lane&15 = column; lanes 0-15 K{0..7,16..23}; lanes 16-31 K{8..15,24..31}
        bf16x16 bfr[4];
#pragma unroll
        for (int s = 0; s < 4; ++s) {
            const __bf16* bp = Bt + (size_t)(nbase + s * 16 + l15) * Ka + kb + half * 8;
            bfr[s] = load_frag2(bp, bp + 16);
        }
        // ---- 8 WMMAs on the loaded fragments ----
#pragma unroll
        for (int s = 0; s < 4; ++s)
            acc[0][s] = __builtin_amdgcn_wmma_f32_16x16x32_bf16(
                false, a0, false, bfr[s], (short)0, acc[0][s], false, false);
#pragma unroll
        for (int s = 0; s < 4; ++s)
            acc[1][s] = __builtin_amdgcn_wmma_f32_16x16x32_bf16(
                false, a1, false, bfr[s], (short)0, acc[1][s], false, false);
    }

    // C/D layout: VGPR r -> M = r + 8*half ; lane&15 -> N
#pragma unroll
    for (int g = 0; g < 2; ++g) {
#pragma unroll
        for (int r = 0; r < 8; ++r) {
            const int m = mt * 32 + g * 16 + half * 8 + r;
            float p0 = 0.f, p1 = 0.f, p2 = 0.f;
            if (posW) { p0 = pos[m * 3 + 0]; p1 = pos[m * 3 + 1]; p2 = pos[m * 3 + 2]; }
            float* crow = C + (size_t)m * Ncols;
#pragma unroll
            for (int s = 0; s < 4; ++s) {
                const int n = nbase + s * 16 + l15;
                float val = acc[g][s][r];
                if (posW) val += p0 * posW[n] + p1 * posW[Ncols + n] + p2 * posW[2 * Ncols + n];
                crow[n] = val;
            }
        }
    }
}

// ---------------- GCN aggregation (deg==7 uniformly) + bias + BN(eval) + ReLU ----------------
__global__ __launch_bounds__(256)
void agg_bn_relu(const float* __restrict__ hw, const int* __restrict__ src,
                 const float* __restrict__ bias,
                 const float* __restrict__ g, const float* __restrict__ be,
                 const float* __restrict__ m, const float* __restrict__ v,
                 __bf16* __restrict__ out_bf, float* __restrict__ out_f32) {
    const int i = blockIdx.x, c = threadIdx.x;
    float acc = hw[(size_t)i * Hf + c];
#pragma unroll
    for (int k = 0; k < Kn; ++k) {
        int s = src[(size_t)i * Kn + k];
        acc += hw[(size_t)s * Hf + c];
    }
    acc *= (1.0f / 7.0f);
    acc += bias[c];
    acc = (acc - m[c]) * (g[c] * rsqrtf(v[c] + EPS_BN)) + be[c];
    acc = fmaxf(acc, 0.0f);
    if (out_bf)  out_bf[(size_t)i * Hf + c]  = f2bf(acc);
    if (out_f32) out_f32[(size_t)i * Hf + c] = acc;
}

// ---------------- attention pooling + heads (one block per graph) ----------------
__global__ __launch_bounds__(256)
void pool_kernel(const float* __restrict__ h3,
                 const float* __restrict__ gate_w, const float* __restrict__ gate_b,
                 const float* __restrict__ aff_w,  const float* __restrict__ aff_b,
                 const float* __restrict__ rl_w1,  const float* __restrict__ rl_b1,
                 const float* __restrict__ rl_w2,  const float* __restrict__ rl_b2,
                 float* __restrict__ out_aff, float* __restrict__ out_pose) {
    __shared__ float sg[Nn];        // gate -> attention weights
    __shared__ float red[Hf];
    __shared__ float gw[Hf];
    __shared__ float gemb[Hf];
    __shared__ float tbuf[Hf];
    const int b = blockIdx.x, tid = threadIdx.x;
    const int base = b * Nn;

    gw[tid] = gate_w[tid];
    __syncthreads();
    const float gb0 = gate_b[0];

    // gate[n] = relu(h3[n] . gate_w + gate_b)
    for (int n = tid; n < Nn; n += 256) {
        const float* row = h3 + (size_t)(base + n) * Hf;
        float s = 0.f;
        for (int c = 0; c < Hf; ++c) s += row[c] * gw[c];
        sg[n] = fmaxf(s + gb0, 0.0f);
    }
    __syncthreads();

    // softmax over the 512 nodes of this graph
    red[tid] = fmaxf(sg[tid], sg[tid + 256]);
    __syncthreads();
    for (int s = 128; s > 0; s >>= 1) { if (tid < s) red[tid] = fmaxf(red[tid], red[tid + s]); __syncthreads(); }
    const float mx = red[0];
    __syncthreads();
    float e0 = __expf(sg[tid] - mx), e1 = __expf(sg[tid + 256] - mx);
    sg[tid] = e0; sg[tid + 256] = e1;
    red[tid] = e0 + e1;
    __syncthreads();
    for (int s = 128; s > 0; s >>= 1) { if (tid < s) red[tid] += red[tid + s]; __syncthreads(); }
    const float inv = 1.0f / red[0];
    __syncthreads();
    sg[tid] *= inv; sg[tid + 256] *= inv;
    __syncthreads();

    // gemb[c] = sum_n attn[n] * h3[n,c]   (coalesced across threads)
    float acc = 0.f;
    for (int n = 0; n < Nn; ++n) acc += sg[n] * h3[(size_t)(base + n) * Hf + tid];
    gemb[tid] = acc;
    __syncthreads();

    // affinity = gemb . aff_w + aff_b
    red[tid] = gemb[tid] * aff_w[tid];
    __syncthreads();
    for (int s = 128; s > 0; s >>= 1) { if (tid < s) red[tid] += red[tid + s]; __syncthreads(); }
    if (tid == 0) out_aff[b] = red[0] + aff_b[0];
    __syncthreads();

    // t = relu(gemb @ rl_w1 + rl_b1)
    float s1 = rl_b1[tid];
    for (int c = 0; c < Hf; ++c) s1 += gemb[c] * rl_w1[c * Hf + tid];
    tbuf[tid] = fmaxf(s1, 0.0f);
    __syncthreads();

    // pose = t @ rl_w2 + rl_b2
    for (int d = 0; d < 3; ++d) {
        red[tid] = tbuf[tid] * rl_w2[tid * 3 + d];
        __syncthreads();
        for (int s = 128; s > 0; s >>= 1) { if (tid < s) red[tid] += red[tid + s]; __syncthreads(); }
        if (tid == 0) out_pose[b * 3 + d] = red[0] + rl_b2[d];
        __syncthreads();
    }
}

// ---------------- launch ----------------
extern "C" void kernel_launch(void* const* d_in, const int* in_sizes, int n_in,
                              void* d_out, int out_size, void* d_ws, size_t ws_size,
                              hipStream_t stream) {
    const float* x      = (const float*)d_in[0];
    const float* pos    = (const float*)d_in[1];
    const float* W1     = (const float*)d_in[2];  const float* b1 = (const float*)d_in[3];
    const float* W2     = (const float*)d_in[4];  const float* b2 = (const float*)d_in[5];
    const float* W3     = (const float*)d_in[6];  const float* b3 = (const float*)d_in[7];
    const float* g1  = (const float*)d_in[8];  const float* be1 = (const float*)d_in[9];
    const float* m1  = (const float*)d_in[10]; const float* v1  = (const float*)d_in[11];
    const float* g2  = (const float*)d_in[12]; const float* be2 = (const float*)d_in[13];
    const float* m2  = (const float*)d_in[14]; const float* v2  = (const float*)d_in[15];
    const float* g3  = (const float*)d_in[16]; const float* be3 = (const float*)d_in[17];
    const float* m3  = (const float*)d_in[18]; const float* v3  = (const float*)d_in[19];
    const float* gate_w = (const float*)d_in[20]; const float* gate_b = (const float*)d_in[21];
    const float* aff_w  = (const float*)d_in[22]; const float* aff_b  = (const float*)d_in[23];
    const float* rl_w1  = (const float*)d_in[24]; const float* rl_b1  = (const float*)d_in[25];
    const float* rl_w2  = (const float*)d_in[26]; const float* rl_b2  = (const float*)d_in[27];

    // workspace carve (256B aligned)
    char* p = (char*)d_ws;
    auto carve = [&](size_t bytes) { void* r = (void*)p; p += (bytes + 255) & ~(size_t)255; return r; };
    __bf16* xbf = (__bf16*)carve((size_t)NBn * INFt * 2);
    __bf16* Wt1 = (__bf16*)carve((size_t)Hf * INFt * 2);
    __bf16* Wt2 = (__bf16*)carve((size_t)Hf * Hf * 2);
    __bf16* Wt3 = (__bf16*)carve((size_t)Hf * Hf * 2);
    int*    src = (int*)   carve((size_t)NBn * Kn * 4);
    float*  hw  = (float*) carve((size_t)NBn * Hf * 4);
    __bf16* hbf = (__bf16*)carve((size_t)NBn * Hf * 2);
    float*  h3  = (float*) carve((size_t)NBn * Hf * 4);

    float* out_aff  = (float*)d_out;          // [128]
    float* out_pose = (float*)d_out + Bg;     // [128*3]

    // precision prep
    cvt_f32_bf16<<<(NBn * INFt + 255) / 256, 256, 0, stream>>>(x, xbf, NBn * INFt);
    cvt_transpose_bf16<<<(INFt * Hf + 255) / 256, 256, 0, stream>>>(W1, Wt1, INFt, Hf); // first 128 rows of W1
    cvt_transpose_bf16<<<(Hf * Hf + 255) / 256, 256, 0, stream>>>(W2, Wt2, Hf, Hf);
    cvt_transpose_bf16<<<(Hf * Hf + 255) / 256, 256, 0, stream>>>(W3, Wt3, Hf, Hf);

    // graph construction
    knn_kernel<<<Bg, Nn, 0, stream>>>(pos, src);

    // wave-tiles: (M/32) * (Ncols/64), 8 waves per block
    const int gemm_blocks = (NBn / 32) * (Hf / 64) / 8;   // 1024

    // layer 1: hw = x@W1[:128] (+ pos@W1[128:131] in epilogue)
    gemm_bf16_wmma<<<gemm_blocks, 256, 0, stream>>>(xbf, Wt1, hw, NBn, INFt, Hf,
                                                    W1 + (size_t)INFt * Hf, pos);
    agg_bn_relu<<<NBn, Hf, 0, stream>>>(hw, src, b1, g1, be1, m1, v1, hbf, nullptr);

    // layer 2
    gemm_bf16_wmma<<<gemm_blocks, 256, 0, stream>>>(hbf, Wt2, hw, NBn, Hf, Hf, nullptr, nullptr);
    agg_bn_relu<<<NBn, Hf, 0, stream>>>(hw, src, b2, g2, be2, m2, v2, hbf, nullptr);

    // layer 3 (fp32 output for pooling)
    gemm_bf16_wmma<<<gemm_blocks, 256, 0, stream>>>(hbf, Wt3, hw, NBn, Hf, Hf, nullptr, nullptr);
    agg_bn_relu<<<NBn, Hf, 0, stream>>>(hw, src, b3, g3, be3, m3, v3, nullptr, h3);

    // pooling + heads
    pool_kernel<<<Bg, Hf, 0, stream>>>(h3, gate_w, gate_b, aff_w, aff_b,
                                       rl_w1, rl_b1, rl_w2, rl_b2, out_aff, out_pose);

    (void)in_sizes; (void)n_in; (void)out_size; (void)ws_size;
}
